// HeterogeneousMoEFFN_14431090115243
// MI455X (gfx1250) — matrix operand order repaired and checked by reading, not compiled, not run on hardware
//
#include <hip/hip_runtime.h>
#include <hip/hip_bf16.h>
#include <math.h>

typedef __attribute__((ext_vector_type(16))) _Float16 v16h;
typedef __attribute__((ext_vector_type(8)))  float    v8f;
typedef int v4i __attribute__((ext_vector_type(4)));

#define AS1 __attribute__((address_space(1)))
#define AS3 __attribute__((address_space(3)))

#if __has_builtin(__builtin_amdgcn_global_load_async_to_lds_b128) && \
    __has_builtin(__builtin_amdgcn_s_wait_asynccnt)
#define USE_ASYNC_LDS 1
#else
#define USE_ASYNC_LDS 0
#endif

#define BB   8
#define SS   2048
#define DD   512
#define EE   6
#define FFN  1024
#define FREQ 1025          // S/2 + 1
#define NTOK (BB * SS)     // 16384

// ---------------------------------------------------------------------------
// small utility kernels
// ---------------------------------------------------------------------------
__global__ void init_accum_kernel(float* accum) {
  if (threadIdx.x < 16) accum[threadIdx.x] = 0.0f;
}

__global__ void convert_f16_kernel(const float* __restrict__ src,
                                   _Float16* __restrict__ dst, int n) {
  int i = blockIdx.x * blockDim.x + threadIdx.x;
  if (i < n) dst[i] = (_Float16)src[i];
}

// ---------------------------------------------------------------------------
// router: one wave (32 lanes) per token
// ---------------------------------------------------------------------------
__global__ __launch_bounds__(256)
void router_kernel(const float* __restrict__ x, const float* __restrict__ rw,
                   const float* __restrict__ rb, const float* __restrict__ gw,
                   const float* __restrict__ gb, float* __restrict__ gate_e,
                   float* __restrict__ gsh, float* __restrict__ accum) {
  __shared__ float sload[EE], simp[EE];
  if (threadIdx.x < EE) { sload[threadIdx.x] = 0.0f; simp[threadIdx.x] = 0.0f; }
  __syncthreads();

  int lane = threadIdx.x & 31;
  int wid  = threadIdx.x >> 5;
  int tok  = blockIdx.x * 8 + wid;          // 2048 blocks * 8 waves = 16384

  float acc[EE] = {0.f, 0.f, 0.f, 0.f, 0.f, 0.f};
  float gacc = 0.f;
  for (int d = lane; d < DD; d += 32) {
    float xv = x[(size_t)tok * DD + d];
#pragma unroll
    for (int e = 0; e < EE; e++) acc[e] += xv * rw[d * EE + e];
    gacc += xv * gw[d];
  }
#pragma unroll
  for (int off = 16; off; off >>= 1) {
#pragma unroll
    for (int e = 0; e < EE; e++) acc[e] += __shfl_xor(acc[e], off, 32);
    gacc += __shfl_xor(gacc, off, 32);
  }

  if (lane == 0) {
    float lg[EE], p[EE];
    float mx = -3.4e38f;
#pragma unroll
    for (int e = 0; e < EE; e++) { lg[e] = acc[e] + rb[e]; mx = fmaxf(mx, lg[e]); }
    float sum = 0.f;
#pragma unroll
    for (int e = 0; e < EE; e++) { p[e] = __expf(lg[e] - mx); sum += p[e]; }
    float inv = 1.0f / sum;
#pragma unroll
    for (int e = 0; e < EE; e++) p[e] *= inv;
    // top-2 (strict > keeps lowest index on ties, matching lax.top_k)
    int i0 = 0;
#pragma unroll
    for (int e = 1; e < EE; e++) if (p[e] > p[i0]) i0 = e;
    int i1 = (i0 == 0) ? 1 : 0;
#pragma unroll
    for (int e = 0; e < EE; e++) if (e != i0 && p[e] > p[i1]) i1 = e;
    float wsum = p[i0] + p[i1];
#pragma unroll
    for (int e = 0; e < EE; e++) gate_e[(size_t)tok * EE + e] = 0.0f;
    gate_e[(size_t)tok * EE + i0] = p[i0] / wsum;
    gate_e[(size_t)tok * EE + i1] = p[i1] / wsum;
    gsh[tok] = 1.0f / (1.0f + __expf(-(gacc + gb[0])));
    atomicAdd(&sload[i0], 1.0f);
    atomicAdd(&sload[i1], 1.0f);
#pragma unroll
    for (int e = 0; e < EE; e++) atomicAdd(&simp[e], p[e]);
  }
  __syncthreads();
  if (threadIdx.x < EE) {
    atomicAdd(&accum[threadIdx.x],      sload[threadIdx.x]);
    atomicAdd(&accum[EE + threadIdx.x], simp[threadIdx.x]);
  }
}

__global__ void aux_finalize_kernel(const float* __restrict__ accum,
                                    float* __restrict__ aux_out) {
  float inv = 1.0f / (float)NTOK;
  float s = 0.f;
#pragma unroll
  for (int e = 0; e < EE; e++) s += (accum[e] * inv) * (accum[EE + e] * inv);
  *aux_out = (float)EE * s;
}

// ---------------------------------------------------------------------------
// depthwise conv (KS=3, SAME) + residual, output f16 (GEMM-ready)
// ---------------------------------------------------------------------------
__global__ __launch_bounds__(256)
void conv_kernel(const float* __restrict__ x, const float* __restrict__ k3,
                 _Float16* __restrict__ outh) {
  int idx = blockIdx.x * blockDim.x + threadIdx.x;   // < B*S*D
  int d = idx % DD;
  int s = (idx / DD) % SS;
  int b = idx / (DD * SS);
  float c = 0.f;
#pragma unroll
  for (int j = 0; j < 3; j++) {
    int sp = s + j - 1;
    if (sp >= 0 && sp < SS)
      c += x[((size_t)b * SS + sp) * DD + d] * k3[j * DD + d];
  }
  outh[idx] = (_Float16)(x[idx] + c);
}

// ---------------------------------------------------------------------------
// Stockham radix-2 FFT, N=2048, in LDS (natural order output)
// ---------------------------------------------------------------------------
__device__ float2* fft2048_lds(float2* src, float2* dst, float sgn) {
  const int N = 2048;
  for (int Ns = 1; Ns < N; Ns <<= 1) {
    for (int j = threadIdx.x; j < N / 2; j += blockDim.x) {
      int k = j & (Ns - 1);
      float ang = sgn * 3.14159265358979f * (float)k / (float)Ns;
      float sn, cs; __sincosf(ang, &sn, &cs);
      float2 u = src[j];
      float2 t = src[j + N / 2];
      float2 v = make_float2(cs * t.x - sn * t.y, cs * t.y + sn * t.x);
      int id = ((j - k) << 1) + k;      // (j/Ns)*2Ns + k
      dst[id]      = make_float2(u.x + v.x, u.y + v.y);
      dst[id + Ns] = make_float2(u.x - v.x, u.y - v.y);
    }
    __syncthreads();
    float2* tmp = src; src = dst; dst = tmp;
  }
  return src;   // 11 stages (odd) -> result lands in the second buffer
}

__global__ __launch_bounds__(256)
void fft_forward_kernel(const float* __restrict__ x, _Float16* __restrict__ cat) {
  __shared__ float2 buf0[2048];
  __shared__ float2 buf1[2048];
  int b = blockIdx.x / DD;
  int d = blockIdx.x % DD;
  for (int s = threadIdx.x; s < SS; s += blockDim.x)
    buf0[s] = make_float2(x[((size_t)b * SS + s) * DD + d], 0.0f);
  __syncthreads();
  float2* res = fft2048_lds(buf0, buf1, -1.0f);
  for (int f = threadIdx.x; f < FREQ; f += blockDim.x) {
    size_t row = (size_t)b * FREQ + f;
    cat[row * (2 * DD) + d]      = (_Float16)res[f].x;
    cat[row * (2 * DD) + DD + d] = (_Float16)res[f].y;
  }
}

__global__ __launch_bounds__(256)
void fft_inverse_accum_kernel(const float* __restrict__ ofreq,
                              const float* __restrict__ gate_e, int ecol,
                              float* __restrict__ out) {
  __shared__ float2 buf0[2048];
  __shared__ float2 buf1[2048];
  int b = blockIdx.x / DD;
  int d = blockIdx.x % DD;
  // hermitian extension of the half spectrum
  for (int f = threadIdx.x; f < SS; f += blockDim.x) {
    float re, im;
    if (f < FREQ) {
      size_t row = (size_t)b * FREQ + f;
      re = ofreq[row * (2 * DD) + d];
      im = ofreq[row * (2 * DD) + DD + d];
    } else {
      size_t row = (size_t)b * FREQ + (SS - f);
      re =  ofreq[row * (2 * DD) + d];
      im = -ofreq[row * (2 * DD) + DD + d];
    }
    buf0[f] = make_float2(re, im);
  }
  __syncthreads();
  float2* res = fft2048_lds(buf0, buf1, +1.0f);
  const float invN = 1.0f / (float)SS;
  for (int s = threadIdx.x; s < SS; s += blockDim.x) {
    size_t tok = (size_t)b * SS + s;
    float g = gate_e[tok * EE + ecol];
    out[tok * DD + d] += g * res[s].x * invN;
  }
}

// ---------------------------------------------------------------------------
// WMMA GEMM: C[M,N] = epilogue(A_f16[M,Kd] @ W_f32[Kd,N] + bias[N])
// 128x128 tile / block, 8 waves (4x2), 32x64 per wave via 2x4 wmma 16x16x32 f16
// Double-buffered LDS, software-pipelined: next tile's global loads issue
// before the WMMA block, conversion/stores after, one barrier per K-step.
// A tile goes global->LDS via async copy (no VGPR round trip) when available.
// 40-half row pitch (80B) gives conflict-free 16-lane ds_load_b128.
// ---------------------------------------------------------------------------
#define TM  128
#define TN  128
#define TK  32
#define TKP 40                       // padded row pitch in halves (80 bytes)
enum { EP_GELU_F16 = 0, EP_SHARED = 1, EP_ACCUM = 2, EP_STORE = 3 };

union H2Pack { _Float16 h[2]; unsigned int u; };

__global__ __launch_bounds__(256)
void gemm_wmma_kernel(const _Float16* __restrict__ A, const float* __restrict__ Bw,
                      const float* __restrict__ bias, int M, int Kd, int N,
                      int mode, const float* __restrict__ gate, int gcol,
                      _Float16* __restrict__ outH, float* __restrict__ outF) {
  __shared__ __align__(16) _Float16 As[2][TM][TKP];   // [buf][m][k]
  __shared__ __align__(16) _Float16 Bs[2][TN][TKP];   // [buf][n][k] (transposed)

  const int tid  = threadIdx.x;
  const int lane = tid & 31;
  const int wid  = tid >> 5;
  const int wm   = wid >> 1;            // 0..3 -> 32 rows each
  const int wn   = wid & 1;             // 0..1 -> 64 cols each
  const int half = lane >> 4;           // wave32 half
  const int l16  = lane & 15;
  const int mBase = blockIdx.y * TM;
  const int nBase = blockIdx.x * TN;

  v8f acc[2][4] = {};

  // A staging: 2 threads/row, 16 contiguous halves each
  const int arow = tid >> 1;
  const int acol = (tid & 1) * 16;
  const int agrow = mBase + arow;
  const bool arow_ok = agrow < M;
  // B staging: thread handles 2 consecutive k-rows x 8 columns
  const int bkp = (tid >> 4) * 2;       // k pair base: 0,2,..,30
  const int bcg = (tid & 15) * 8;       // column group: 0,8,..,120

  // ---- prologue: synchronous stage of tile 0 into buffer 0 ----
  {
    if (arow_ok) {
      const _Float16* src = A + (size_t)agrow * Kd + acol;
      *(uint4*)(&As[0][arow][acol])     = *(const uint4*)(src);
      *(uint4*)(&As[0][arow][acol + 8]) = *(const uint4*)(src + 8);
    } else {
      uint4 z = {0u, 0u, 0u, 0u};
      *(uint4*)(&As[0][arow][acol])     = z;
      *(uint4*)(&As[0][arow][acol + 8]) = z;
    }
    const float* r0 = Bw + (size_t)bkp * N + nBase + bcg;
    const float* r1 = r0 + N;
    float4 a0 = *(const float4*)(r0);
    float4 a1 = *(const float4*)(r0 + 4);
    float4 b0 = *(const float4*)(r1);
    float4 b1 = *(const float4*)(r1 + 4);
    float k0[8] = {a0.x, a0.y, a0.z, a0.w, a1.x, a1.y, a1.z, a1.w};
    float k1[8] = {b0.x, b0.y, b0.z, b0.w, b1.x, b1.y, b1.z, b1.w};
#pragma unroll
    for (int j = 0; j < 8; j++) {
      H2Pack p;
      p.h[0] = (_Float16)k0[j];
      p.h[1] = (_Float16)k1[j];
      *(unsigned int*)(&Bs[0][bcg + j][bkp]) = p.u;
    }
  }
  __syncthreads();

  int cur = 0;
  for (int kt = 0; kt < Kd; kt += TK, cur ^= 1) {
    const int nxt = cur ^ 1;
    const bool havenext = (kt + TK) < Kd;
    float4 f0, f1, f2, f3;
    uint4 a0q = {0u, 0u, 0u, 0u}, a1q = {0u, 0u, 0u, 0u};

    // ---- issue next tile's global traffic before computing ----
    if (havenext) {
      const float* r0 = Bw + (size_t)(kt + TK + bkp) * N + nBase + bcg;
      const float* r1 = r0 + N;
      f0 = *(const float4*)(r0);
      f1 = *(const float4*)(r0 + 4);
      f2 = *(const float4*)(r1);
      f3 = *(const float4*)(r1 + 4);
      if (arow_ok) {
        const _Float16* asrc = A + (size_t)agrow * Kd + (kt + TK) + acol;
#if USE_ASYNC_LDS
        __builtin_amdgcn_global_load_async_to_lds_b128(
            (v4i*)(asrc), (AS3 v4i*)(&As[nxt][arow][acol]), 0, 0);
        __builtin_amdgcn_global_load_async_to_lds_b128(
            (v4i*)(asrc + 8), (AS3 v4i*)(&As[nxt][arow][acol + 8]), 0, 0);
#else
        a0q = *(const uint4*)(asrc);
        a1q = *(const uint4*)(asrc + 8);
#endif
      }
    }

    // ---- fragments: all contiguous LDS reads ----
    v16h afrag[2], bfrag[4];
#pragma unroll
    for (int ti = 0; ti < 2; ti++) {
      int r = wm * 32 + ti * 16 + l16;
      *(uint4*)(&afrag[ti])       = *(const uint4*)(&As[cur][r][half * 8]);
      *((uint4*)(&afrag[ti]) + 1) = *(const uint4*)(&As[cur][r][16 + half * 8]);
    }
#pragma unroll
    for (int tj = 0; tj < 4; tj++) {
      int c = wn * 64 + tj * 16 + l16;
      *(uint4*)(&bfrag[tj])       = *(const uint4*)(&Bs[cur][c][half * 8]);
      *((uint4*)(&bfrag[tj]) + 1) = *(const uint4*)(&Bs[cur][c][16 + half * 8]);
    }
#pragma unroll
    for (int ti = 0; ti < 2; ti++)
#pragma unroll
      for (int tj = 0; tj < 4; tj++)
        acc[ti][tj] = __builtin_amdgcn_wmma_f32_16x16x32_f16(
            false, afrag[ti], false, bfrag[tj], (short)0, acc[ti][tj],
            false, false);

    // ---- finish staging next tile ----
    if (havenext) {
      float k0[8] = {f0.x, f0.y, f0.z, f0.w, f1.x, f1.y, f1.z, f1.w};
      float k1[8] = {f2.x, f2.y, f2.z, f2.w, f3.x, f3.y, f3.z, f3.w};
#pragma unroll
      for (int j = 0; j < 8; j++) {
        H2Pack p;
        p.h[0] = (_Float16)k0[j];
        p.h[1] = (_Float16)k1[j];
        *(unsigned int*)(&Bs[nxt][bcg + j][bkp]) = p.u;
      }
#if USE_ASYNC_LDS
      if (!arow_ok) {
        uint4 z = {0u, 0u, 0u, 0u};
        *(uint4*)(&As[nxt][arow][acol])     = z;
        *(uint4*)(&As[nxt][arow][acol + 8]) = z;
      }
      __builtin_amdgcn_s_wait_asynccnt(0);
#else
      *(uint4*)(&As[nxt][arow][acol])     = a0q;
      *(uint4*)(&As[nxt][arow][acol + 8]) = a1q;
#endif
    }
    __syncthreads();
  }

  // epilogue (C/D layout: col = l16, row = half*8 + i)
#pragma unroll
  for (int ti = 0; ti < 2; ti++) {
#pragma unroll
    for (int tj = 0; tj < 4; tj++) {
      int cGlob = nBase + wn * 64 + tj * 16 + l16;
      float bv = bias[cGlob];
#pragma unroll
      for (int i = 0; i < 8; i++) {
        int rGlob = mBase + wm * 32 + ti * 16 + half * 8 + i;
        if (rGlob >= M) continue;
        float v = acc[ti][tj][i] + bv;
        size_t oi = (size_t)rGlob * N + cGlob;
        if (mode == EP_GELU_F16) {
          outH[oi] = (_Float16)(0.5f * v * (1.0f + erff(v * 0.70710678118f)));
        } else if (mode == EP_SHARED) {
          outF[oi] = v * gate[rGlob];
        } else if (mode == EP_ACCUM) {
          outF[oi] += v * gate[(size_t)rGlob * EE + gcol];
        } else { // EP_STORE
          outF[oi] = v;
        }
      }
    }
  }
}

// ---------------------------------------------------------------------------
// launch
// ---------------------------------------------------------------------------
static inline void launch_gemm(const _Float16* A, const float* W, const float* b,
                               int M, int Kd, int N, int mode, const float* gate,
                               int gcol, _Float16* outH, float* outF,
                               hipStream_t stream) {
  dim3 grid(N / TN, (M + TM - 1) / TM);
  gemm_wmma_kernel<<<grid, 256, 0, stream>>>(A, W, b, M, Kd, N, mode, gate,
                                             gcol, outH, outF);
}

extern "C" void kernel_launch(void* const* d_in, const int* in_sizes, int n_in,
                              void* d_out, int out_size, void* d_ws, size_t ws_size,
                              hipStream_t stream) {
  (void)in_sizes; (void)n_in; (void)out_size; (void)ws_size;

  const float* x          = (const float*)d_in[0];
  const float* shared_w1  = (const float*)d_in[1];
  const float* shared_b1  = (const float*)d_in[2];
  const float* shared_w2  = (const float*)d_in[3];
  const float* shared_b2  = (const float*)d_in[4];
  const float* gate_w     = (const float*)d_in[5];
  const float* gate_b     = (const float*)d_in[6];
  const float* router_w   = (const float*)d_in[7];
  const float* router_b   = (const float*)d_in[8];
  const float* conv_k     = (const float*)d_in[9];
  const float* conv_w1    = (const float*)d_in[10];
  const float* conv_b1    = (const float*)d_in[11];
  const float* conv_w2    = (const float*)d_in[12];
  const float* conv_b2    = (const float*)d_in[13];
  const float* four_w1    = (const float*)d_in[14];
  const float* four_b1    = (const float*)d_in[15];
  const float* four_w2    = (const float*)d_in[16];
  const float* four_b2    = (const float*)d_in[17];
  const float* mlp_w1     = (const float*)d_in[18];
  const float* mlp_b1     = (const float*)d_in[19];
  const float* mlp_w2     = (const float*)d_in[20];
  const float* mlp_b2     = (const float*)d_in[21];

  float* out = (float*)d_out;                        // [B*S*D] + aux at end
  const int MF = BB * FREQ;                          // 8200 fourier rows

  // workspace layout
  size_t off = 0;
  auto wsa = [&](size_t bytes) {
    void* p = (char*)d_ws + off;
    off += (bytes + 255) & ~(size_t)255;
    return p;
  };
  _Float16* xh    = (_Float16*)wsa((size_t)NTOK * DD * 2);
  _Float16* cxh   = (_Float16*)wsa((size_t)NTOK * DD * 2);
  _Float16* cath  = (_Float16*)wsa((size_t)8320 * (2 * DD) * 2);
  _Float16* hbuf  = (_Float16*)wsa((size_t)NTOK * FFN * 2);
  float*    ofreq = (float*)wsa((size_t)8320 * (2 * DD) * 4);
  float*    gateE = (float*)wsa((size_t)NTOK * EE * 4);
  float*    gsh   = (float*)wsa((size_t)NTOK * 4);
  float*    accum = (float*)wsa(64);

  // router + aux
  init_accum_kernel<<<1, 32, 0, stream>>>(accum);
  router_kernel<<<NTOK / 8, 256, 0, stream>>>(x, router_w, router_b, gate_w,
                                              gate_b, gateE, gsh, accum);
  aux_finalize_kernel<<<1, 1, 0, stream>>>(accum, out + (size_t)NTOK * DD);

  // f16 copy of x (GEMM A operand for shared/mlp experts)
  convert_f16_kernel<<<(NTOK * DD) / 256, 256, 0, stream>>>(x, xh, NTOK * DD);

  // shared expert (writes '=' into out, initializing it everywhere)
  launch_gemm(xh, shared_w1, shared_b1, NTOK, DD, FFN, EP_GELU_F16, nullptr, 0,
              hbuf, nullptr, stream);
  launch_gemm(hbuf, shared_w2, shared_b2, NTOK, FFN, DD, EP_SHARED, gsh, 0,
              nullptr, out, stream);

  int ci = 0, fi = 0, mi = 0;
  const int types[EE] = {0, 1, 2, 0, 1, 2};  // conv, fourier, mlp, ...
  for (int e = 0; e < EE; e++) {
    if (types[e] == 0) {                     // conv expert
      conv_kernel<<<(NTOK * DD) / 256, 256, 0, stream>>>(
          x, conv_k + (size_t)ci * 3 * DD, cxh);
      launch_gemm(cxh, conv_w1 + (size_t)ci * DD * FFN, conv_b1 + ci * FFN,
                  NTOK, DD, FFN, EP_GELU_F16, nullptr, 0, hbuf, nullptr, stream);
      launch_gemm(hbuf, conv_w2 + (size_t)ci * FFN * DD, conv_b2 + ci * DD,
                  NTOK, FFN, DD, EP_ACCUM, gateE, e, nullptr, out, stream);
      ci++;
    } else if (types[e] == 1) {              // fourier expert
      fft_forward_kernel<<<BB * DD, 256, 0, stream>>>(x, cath);
      launch_gemm(cath, four_w1 + (size_t)fi * 2 * DD * FFN, four_b1 + fi * FFN,
                  MF, 2 * DD, FFN, EP_GELU_F16, nullptr, 0, hbuf, nullptr, stream);
      launch_gemm(hbuf, four_w2 + (size_t)fi * FFN * 2 * DD, four_b2 + fi * 2 * DD,
                  MF, FFN, 2 * DD, EP_STORE, nullptr, 0, nullptr, ofreq, stream);
      fft_inverse_accum_kernel<<<BB * DD, 256, 0, stream>>>(ofreq, gateE, e, out);
      fi++;
    } else {                                 // mlp expert
      launch_gemm(xh, mlp_w1 + (size_t)mi * DD * FFN, mlp_b1 + mi * FFN,
                  NTOK, DD, FFN, EP_GELU_F16, nullptr, 0, hbuf, nullptr, stream);
      launch_gemm(hbuf, mlp_w2 + (size_t)mi * FFN * DD, mlp_b2 + mi * DD,
                  NTOK, FFN, DD, EP_ACCUM, gateE, e, nullptr, out, stream);
      mi++;
    }
  }
}
